// MotionDecoder_87720412054213
// MI455X (gfx1250) — matrix operand order repaired and checked by reading, not compiled, not run on hardware
//
#include <hip/hip_runtime.h>

// ---------------------------------------------------------------------------
// LSTM motion decoder on gfx1250: each wave32 owns a 16-row batch tile and
// runs the full 50-step recurrence with v_wmma_f32_16x16x32_f16.
//   gates(16x256) = [h(64) | inp(2) | 1 | pad(29)](16x96)  @  Bext(96x256)
// Bias is folded in via the ones-column. fc head = 2 more WMMA (N padded to 16).
// ---------------------------------------------------------------------------

typedef __attribute__((ext_vector_type(16))) _Float16 v16h;
typedef __attribute__((ext_vector_type(8)))  _Float16 v8h;
typedef __attribute__((ext_vector_type(8)))  float    v8f;

#define HID       64
#define NT        16    // N-tiles of the gate GEMM (256 / 16)
#define KS        3     // K-slabs (96 / 32)
#define TSTEPS    50
#define WAVES     4     // waves per block
#define ROWSTRIDE 72    // halves per h row in LDS (64 + 8 pad, keeps 16B align)

#define WMMA_F16(A, Bm, C) \
  __builtin_amdgcn_wmma_f32_16x16x32_f16(false, (A), false, (Bm), (short)0, (C), false, false)

__global__ __launch_bounds__(128) void lstm_decoder_wmma(
    const float* __restrict__ agent_feat,   // (B,64)
    const float* __restrict__ start_pos,    // (B,2)
    const float* __restrict__ W_ih,         // (256,2)
    const float* __restrict__ W_hh,         // (256,64)
    const float* __restrict__ b_ih,         // (256,)
    const float* __restrict__ b_hh,         // (256,)
    const float* __restrict__ W_fc,         // (2,64)
    const float* __restrict__ b_fc,         // (2,)
    float* __restrict__ out)                // (B,50,2)
{
  // B-fragment blobs: element (t,s,lane,i) = Bext[k = s*32 + (lane>=16)*16 + i][n = t*16 + lane%16]
  __shared__ alignas(32) _Float16 Bg[NT * KS * 512];      // 48 KB
  __shared__ alignas(32) _Float16 Bfc[2 * 512];           //  2 KB
  __shared__ alignas(32) _Float16 Hl[WAVES][16 * ROWSTRIDE]; // 9 KB, h row-major f16
  __shared__ float Pl[WAVES][16][2];                      // fed-back positions

  const int tid    = threadIdx.x;
  const int lane   = tid & 31;
  const int wid    = tid >> 5;
  const int nL     = lane & 15;   // column within a 16-wide tile / row for A loads
  const int hiHalf = lane >> 4;   // lane-half selects K sub-range
  const int base   = (blockIdx.x * WAVES + wid) * 16;     // batch-row base of this wave

  // ---- stage gate weights as f16 WMMA-B fragments (block-cooperative) ----
  for (int e = tid; e < NT * KS * 512; e += 128) {
    int t = e / (KS * 512);
    int r = e % (KS * 512);
    int s = r / 512;
    int q = r % 512;
    int l = q >> 4;
    int i = q & 15;
    int n = t * 16 + (l & 15);
    int k = s * 32 + ((l >> 4) << 4) + i;   // extended-K index 0..95
    float v;
    if (k < 64)       v = W_hh[n * 64 + k];
    else if (k < 66)  v = W_ih[n * 2 + (k - 64)];
    else if (k == 66) v = b_ih[n] + b_hh[n];     // ones-column folds the bias in
    else              v = 0.0f;
    Bg[e] = (_Float16)v;
  }
  for (int e = tid; e < 2 * 512; e += 128) {
    int s = e / 512;
    int q = e % 512;
    int l = q >> 4;
    int i = q & 15;
    int n = l & 15;
    int k = s * 32 + ((l >> 4) << 4) + i;
    Bfc[e] = (n < 2) ? (_Float16)W_fc[n * 64 + k] : (_Float16)0.0f;
  }

  // ---- stage h0 = agent_feat tile (f32 -> f16, row-major) ----
  for (int i = lane; i < 16 * HID; i += 32) {
    int rrow = i >> 6;
    int col  = i & 63;
    Hl[wid][rrow * ROWSTRIDE + col] = (_Float16)agent_feat[(base + rrow) * HID + col];
  }
  // ---- stage start_pos tile ----
  {
    int rrow = lane >> 1, col = lane & 1;
    Pl[wid][rrow][col] = start_pos[(base + rrow) * 2 + col];
  }
  __syncthreads();

  const float bf0 = b_fc[0], bf1 = b_fc[1];

  // A-fragment load from row-major f16 h in LDS (contiguous K order, same as Bg/Bfc):
  // lane holds row nL; lane-half selects K {0..15} vs {16..31}.
  auto loadA = [&](int s) -> v16h {
    const _Float16* p = &Hl[wid][nL * ROWSTRIDE + s * 32 + hiHalf * 16];
    v8h lo = *(const v8h*)p;          // K = s*32 + off + {0..7}
    v8h hi = *(const v8h*)(p + 8);    // K = s*32 + off + {8..15}
    v16h a;
#pragma unroll
    for (int i = 0; i < 8; ++i) { a[i] = lo[i]; a[8 + i] = hi[i]; }
    return a;
  };
  auto loadBg = [&](int t, int s) -> v16h {
    return *(const v16h*)(&Bg[(t * KS + s) * 512 + lane * 16]);
  };
  auto loadBfc = [&](int s) -> v16h {
    return *(const v16h*)(&Bfc[s * 512 + lane * 16]);
  };
  auto sig = [](float x) { return 1.0f / (1.0f + __expf(-x)); };
  auto tnh = [](float x) { return 2.0f / (1.0f + __expf(-2.0f * x)) - 1.0f; };

  v8f cst[4] = {};   // cell state, C layout: cst[tj][r] = c[row r+hiHalf*8][col tj*16+nL]

#pragma unroll 1
  for (int t = 0; t < TSTEPS; ++t) {
    // A slabs 0,1 = current h (held in regs so h LDS can be overwritten below)
    v16h a0 = loadA(0);
    v16h a1 = loadA(1);
    // A slab 2 = [inp0, inp1, 1, 0...]; zero for the upper lane-half (K 80..95)
    v16h a2 = {};
    if (hiHalf == 0) {
      a2[0] = (_Float16)Pl[wid][nL][0];
      a2[1] = (_Float16)Pl[wid][nL][1];
      a2[2] = (_Float16)1.0f;
    }

#pragma unroll
    for (int tj = 0; tj < 4; ++tj) {
      v8f ai = {}, af = {}, ag = {}, ao = {};
      // gate order i,f,g,o -> N-tiles tj, 4+tj, 8+tj, 12+tj (lane-aligned columns)
      ai = WMMA_F16(a0, loadBg(tj,      0), ai);
      ai = WMMA_F16(a1, loadBg(tj,      1), ai);
      ai = WMMA_F16(a2, loadBg(tj,      2), ai);
      af = WMMA_F16(a0, loadBg(4 + tj,  0), af);
      af = WMMA_F16(a1, loadBg(4 + tj,  1), af);
      af = WMMA_F16(a2, loadBg(4 + tj,  2), af);
      ag = WMMA_F16(a0, loadBg(8 + tj,  0), ag);
      ag = WMMA_F16(a1, loadBg(8 + tj,  1), ag);
      ag = WMMA_F16(a2, loadBg(8 + tj,  2), ag);
      ao = WMMA_F16(a0, loadBg(12 + tj, 0), ao);
      ao = WMMA_F16(a1, loadBg(12 + tj, 1), ao);
      ao = WMMA_F16(a2, loadBg(12 + tj, 2), ao);

#pragma unroll
      for (int r = 0; r < 8; ++r) {
        float iv = sig(ai[r]);
        float fv = sig(af[r]);
        float gv = tnh(ag[r]);
        float ov = sig(ao[r]);
        float cn = fv * cst[tj][r] + iv * gv;
        cst[tj][r] = cn;
        float hn = ov * tnh(cn);
        // write h back (C layout -> row-major f16); a0/a1 already in regs
        Hl[wid][(r + hiHalf * 8) * ROWSTRIDE + tj * 16 + nL] = (_Float16)hn;
      }
    }

    // fc head: pos(16x2 in a 16x16 tile) = h_new(16x64) @ Wfc^T, K = 2 slabs of 32
    v16h h0f = loadA(0);
    v16h h1f = loadA(1);
    v8f pf = {};
    pf = WMMA_F16(h0f, loadBfc(0), pf);
    pf = WMMA_F16(h1f, loadBfc(1), pf);

    if (nL < 2) {   // lanes 0,1 carry rows 0..7; lanes 16,17 carry rows 8..15
      float bb = (nL == 0) ? bf0 : bf1;
#pragma unroll
      for (int r = 0; r < 8; ++r) {
        float pv = pf[r] + bb;
        int grow = base + r + hiHalf * 8;
        out[grow * (TSTEPS * 2) + t * 2 + nL] = pv;  // (B,50,2)
        Pl[wid][r + hiHalf * 8][nL] = pv;            // feed back as next input
      }
    }
  }
}

extern "C" void kernel_launch(void* const* d_in, const int* in_sizes, int n_in,
                              void* d_out, int out_size, void* d_ws, size_t ws_size,
                              hipStream_t stream) {
  const float* agent_feat = (const float*)d_in[0];
  const float* start_pos  = (const float*)d_in[1];
  const float* W_ih       = (const float*)d_in[2];
  const float* W_hh       = (const float*)d_in[3];
  const float* b_ih       = (const float*)d_in[4];
  const float* b_hh       = (const float*)d_in[5];
  const float* W_fc       = (const float*)d_in[6];
  const float* b_fc       = (const float*)d_in[7];
  float* out = (float*)d_out;

  const int B = 32768;
  dim3 grid(B / (16 * WAVES));   // 512 blocks x 4 waves x 16 rows = 32768
  dim3 block(128);
  lstm_decoder_wmma<<<grid, block, 0, stream>>>(
      agent_feat, start_pos, W_ih, W_hh, b_ih, b_hh, W_fc, b_fc, out);
}
